// BayesGNN_48335561949570
// MI455X (gfx1250) — matrix operand order repaired and checked by reading, not compiled, not run on hardware
//
#include <hip/hip_runtime.h>
#include <hip/hip_bf16.h>
#include <math.h>

// ---------------------------------------------------------------------------
// BayesGNN forward for MI455X (gfx1250, wave32).
//   * GEMMs: v_wmma_f32_16x16x32_bf16, split-precision bf16 hi/lo operands
//     (3 WMMAs per K-step -> ~fp32 accuracy, f32 accumulation). The fp32->bf16
//     split is hoisted into one-time staging kernels so the GEMM inner loop is
//     pure LDS-load + WMMA (no per-tile cvt VALU).
//   * Tile fills use global_load_async_to_lds_b128 (ASYNCcnt) when available.
//   * Message passing (the real roofline term: ~7 GB gather+atomic traffic)
//     streams coalesced 256-wide feature rows with one norm per edge.
// ---------------------------------------------------------------------------

typedef __attribute__((ext_vector_type(16))) __bf16 v16bf;
typedef __attribute__((ext_vector_type(8)))  float  v8f;
typedef unsigned int u32;
typedef int vi4 __attribute__((vector_size(4 * sizeof(int))));  // builtin's pointee type

#define TM 128    // rows per block  (8 waves x 16)
#define TN 32     // cols per block  (2 wmma tiles per wave)
#define TK 32     // K per main-loop step (one wmma K)
#define LDA (TK + 8)   // LDS row stride in bf16: 80B rows -> 20-uint stride (5 x 16B)
#define HID 256
#define NGRAPH 64

#define AS1 __attribute__((address_space(1)))
#define AS3 __attribute__((address_space(3)))

#if __has_builtin(__builtin_amdgcn_global_load_async_to_lds_b128) && \
    __has_builtin(__builtin_amdgcn_s_wait_asynccnt)
#define ASYNC_LDS 1
#else
#define ASYNC_LDS 0
#endif

__device__ __forceinline__ void cp16(void* lds_dst, const void* gsrc) {
#if ASYNC_LDS
    // direct global -> LDS, tracked with ASYNCcnt (no VGPR round trip)
    __builtin_amdgcn_global_load_async_to_lds_b128(
        (AS1 vi4*)const_cast<void*>(gsrc), (AS3 vi4*)lds_dst, 0, 0);
#else
    *reinterpret_cast<uint4*>(lds_dst) = *reinterpret_cast<const uint4*>(gsrc);
#endif
}

__device__ __forceinline__ void zero16(void* lds_dst) {
    uint4 z; z.x = z.y = z.z = z.w = 0u;
    *reinterpret_cast<uint4*>(lds_dst) = z;
}

__device__ __forceinline__ void wait_async_copies() {
#if ASYNC_LDS
    __builtin_amdgcn_s_wait_asynccnt(0);
#endif
}

union Frag { v16bf bf; uint4 q[2]; };

// ---------------------------------------------------------------------------
// Tiled WMMA GEMM on pre-split bf16 operands.
//   A  : [M][K]  row-major hi/lo bf16
//   Bt : [Nc][K] (B transposed) hi/lo bf16
//   C  : [M][Nc] f32 = act(A@B + bias); act: 0 none, 1 relu, 2 softplus+1e-3
// ---------------------------------------------------------------------------
__global__ __launch_bounds__(256) void wmma_gemm_kernel(
    const __bf16* __restrict__ Ah, const __bf16* __restrict__ Al,
    const __bf16* __restrict__ Bth, const __bf16* __restrict__ Btl,
    const float* __restrict__ bias, float* __restrict__ C,
    int M, int K, int Nc, int act)
{
    __shared__ __align__(16) __bf16 sAh[TM][LDA];
    __shared__ __align__(16) __bf16 sAl[TM][LDA];
    __shared__ __align__(16) __bf16 sBh[TN][LDA];
    __shared__ __align__(16) __bf16 sBl[TN][LDA];

    const int tid  = threadIdx.x;
    const int wave = tid >> 5;           // wave32
    const int lane = tid & 31;

    const int block_row = blockIdx.x * TM;
    const int block_col = blockIdx.y * TN;

    v8f acc[2] = {};

    // Fragment addressing (wave32, 16-bit operands):
    //  A 16x32: lanes 0-15 -> M=lane, K {0..7,16..23}; lanes 16-31 -> K {8..15,24..31}
    //  B 32x16: lanes 0-15 -> N=lane, K {0..7,16..23}; lanes 16-31 -> K {8..15,24..31}
    const int arow = wave * 16 + (lane & 15);
    const int aoff = (lane < 16) ? 0 : 4;     // in uints
    const int boff = (lane < 16) ? 0 : 4;     // in uints

    for (int k0 = 0; k0 < K; k0 += TK) {
        // ---- A tiles: TM rows x 4 chunks of 16B, hi & lo ----
        for (int i = tid; i < TM * 4; i += 256) {
            int r = i >> 2, ch = i & 3;
            int gr = block_row + r;
            void* dh = &sAh[r][ch * 8];
            void* dl = &sAl[r][ch * 8];
            if (gr < M) {
                size_t g = (size_t)gr * K + k0 + ch * 8;
                cp16(dh, Ah + g);
                cp16(dl, Al + g);
            } else { zero16(dh); zero16(dl); }
        }
        // ---- B tiles: TN cols (rows of Bt) x 4 chunks, hi & lo ----
        for (int i = tid; i < TN * 4; i += 256) {
            int r = i >> 2, ch = i & 3;
            int gc = block_col + r;
            void* dh = &sBh[r][ch * 8];
            void* dl = &sBl[r][ch * 8];
            if (gc < Nc) {
                size_t g = (size_t)gc * K + k0 + ch * 8;
                cp16(dh, Bth + g);
                cp16(dl, Btl + g);
            } else { zero16(dh); zero16(dl); }
        }
        wait_async_copies();
        __syncthreads();

        // ---- build fragments: pure b128 LDS loads, no conversions ----
        const u32* pAh = reinterpret_cast<const u32*>(&sAh[arow][0]);
        const u32* pAl = reinterpret_cast<const u32*>(&sAl[arow][0]);
        Frag ah, al;
        ah.q[0] = *reinterpret_cast<const uint4*>(pAh + aoff);
        ah.q[1] = *reinterpret_cast<const uint4*>(pAh + 8 + aoff);
        al.q[0] = *reinterpret_cast<const uint4*>(pAl + aoff);
        al.q[1] = *reinterpret_cast<const uint4*>(pAl + 8 + aoff);

        #pragma unroll
        for (int t = 0; t < 2; ++t) {
            const u32* pBh = reinterpret_cast<const u32*>(&sBh[t * 16 + (lane & 15)][0]);
            const u32* pBl = reinterpret_cast<const u32*>(&sBl[t * 16 + (lane & 15)][0]);
            Frag bh, bl;
            bh.q[0] = *reinterpret_cast<const uint4*>(pBh + boff);
            bh.q[1] = *reinterpret_cast<const uint4*>(pBh + boff + 8);
            bl.q[0] = *reinterpret_cast<const uint4*>(pBl + boff);
            bl.q[1] = *reinterpret_cast<const uint4*>(pBl + boff + 8);

            // hi*hi + hi*lo + lo*hi  -> ~fp32 product, f32 accumulate
            acc[t] = __builtin_amdgcn_wmma_f32_16x16x32_bf16(
                false, ah.bf, false, bh.bf, (short)0, acc[t], false, false);
            acc[t] = __builtin_amdgcn_wmma_f32_16x16x32_bf16(
                false, ah.bf, false, bl.bf, (short)0, acc[t], false, false);
            acc[t] = __builtin_amdgcn_wmma_f32_16x16x32_bf16(
                false, al.bf, false, bh.bf, (short)0, acc[t], false, false);
        }
        __syncthreads();
    }

    // C/D layout (f32 16x16): VGPR r -> M = r (+8 for lanes 16-31), N = lane&15
    const int rbase = block_row + wave * 16 + ((lane < 16) ? 0 : 8);
    #pragma unroll
    for (int t = 0; t < 2; ++t) {
        int gcol = block_col + t * 16 + (lane & 15);
        if (gcol >= Nc) continue;
        float bv = bias ? bias[gcol] : 0.0f;
        #pragma unroll
        for (int r = 0; r < 8; ++r) {
            int grow = rbase + r;
            if (grow < M) {
                float v = acc[t][r] + bv;
                if (act == 1) {
                    v = fmaxf(v, 0.0f);
                } else if (act == 2) {
                    v = fmaxf(v, 0.0f) + log1pf(expf(-fabsf(v))) + 0.001f;
                }
                C[(size_t)grow * Nc + gcol] = v;
            }
        }
    }
}

// ---------------------------------------------------------------------------
// Split-precision staging kernels (fp32 -> bf16 hi + bf16 residual)
// ---------------------------------------------------------------------------
__global__ void split_kernel(const float* __restrict__ src,
                             __bf16* __restrict__ hi, __bf16* __restrict__ lo,
                             int total)
{
    int i = blockIdx.x * blockDim.x + threadIdx.x;
    if (i < total) {
        float v = src[i];
        __bf16 h = (__bf16)v;
        hi[i] = h;
        lo[i] = (__bf16)(v - (float)h);
    }
}

// transpose + split: src [rows][cols] -> dst [cols][rows]
__global__ void split_t_kernel(const float* __restrict__ src,
                               __bf16* __restrict__ hiT, __bf16* __restrict__ loT,
                               int rows, int cols)
{
    int i = blockIdx.x * blockDim.x + threadIdx.x;
    if (i < rows * cols) {
        int r = i / cols, c = i % cols;
        float v = src[i];
        __bf16 h = (__bf16)v;
        size_t o = (size_t)c * rows + r;
        hiT[o] = h;
        loT[o] = (__bf16)(v - (float)h);
    }
}

// ---------------------------------------------------------------------------
// Degree / normalization
// ---------------------------------------------------------------------------
__global__ void init_deg_kernel(float* __restrict__ deg, int n) {
    int i = blockIdx.x * blockDim.x + threadIdx.x;
    if (i < n) deg[i] = 1.0f;                    // self-loop
}
__global__ void count_deg_kernel(const int* __restrict__ dst,
                                 float* __restrict__ deg, int e) {
    int i = blockIdx.x * blockDim.x + threadIdx.x;
    if (i < e) atomicAdd(&deg[dst[i]], 1.0f);
}
__global__ void rsqrt_kernel(float* __restrict__ deg, int n) {
    int i = blockIdx.x * blockDim.x + threadIdx.x;
    if (i < n) deg[i] = rsqrtf(deg[i]);          // in place: deg -> dinv
}

// ---------------------------------------------------------------------------
// Message passing: agg[d,:] = dinv[d]^2*h[d,:] + sum_e dinv[s]dinv[d]*h[s,:]
// ---------------------------------------------------------------------------
__global__ __launch_bounds__(HID) void self_init_kernel(
    const float* __restrict__ dinv, const float* __restrict__ h,
    float* __restrict__ agg)
{
    int i = blockIdx.x;
    float nrm = dinv[i] * dinv[i];
    size_t base = (size_t)i * HID;
    agg[base + threadIdx.x] = nrm * h[base + threadIdx.x];
}

__global__ __launch_bounds__(HID) void scatter_edges_kernel(
    const int* __restrict__ src, const int* __restrict__ dst,
    const float* __restrict__ dinv, const float* __restrict__ h,
    float* __restrict__ agg, int e)
{
    for (int ed = blockIdx.x; ed < e; ed += gridDim.x) {
        int s = src[ed], d = dst[ed];
        float nrm = dinv[s] * dinv[d];
        atomicAdd(&agg[(size_t)d * HID + threadIdx.x],
                  nrm * h[(size_t)s * HID + threadIdx.x]);
    }
}

// out = relu(in + b) kept in fp32 (needed for pooling)
__global__ void bias_relu_kernel(const float* __restrict__ in,
                                 const float* __restrict__ bias,
                                 float* __restrict__ out, int total)
{
    int i = blockIdx.x * blockDim.x + threadIdx.x;
    if (i < total) out[i] = fmaxf(in[i] + bias[i % HID], 0.0f);
}

// fused: relu(in + b) -> bf16 hi/lo split (feeds the next GEMM directly)
__global__ void bias_relu_split_kernel(const float* __restrict__ in,
                                       const float* __restrict__ bias,
                                       __bf16* __restrict__ hi,
                                       __bf16* __restrict__ lo, int total)
{
    int i = blockIdx.x * blockDim.x + threadIdx.x;
    if (i < total) {
        float v = fmaxf(in[i] + bias[i % HID], 0.0f);
        __bf16 h = (__bf16)v;
        hi[i] = h;
        lo[i] = (__bf16)(v - (float)h);
    }
}

// ---------------------------------------------------------------------------
// Global mean pooling
// ---------------------------------------------------------------------------
__global__ void zero_pool_kernel(float* __restrict__ pool,
                                 float* __restrict__ cnt) {
    int i = blockIdx.x * blockDim.x + threadIdx.x;
    if (i < NGRAPH * HID) pool[i] = 0.0f;
    if (i < NGRAPH) cnt[i] = 0.0f;
}

__global__ __launch_bounds__(HID) void pool_accum_kernel(
    const int* __restrict__ batch, const float* __restrict__ h,
    float* __restrict__ pool, float* __restrict__ cnt)
{
    int i = blockIdx.x;
    int g = batch[i];
    atomicAdd(&pool[(size_t)g * HID + threadIdx.x],
              h[(size_t)i * HID + threadIdx.x]);
    if (threadIdx.x == 0) atomicAdd(&cnt[g], 1.0f);
}

// fused: pooled mean -> bf16 hi/lo split (feeds head GEMM)
__global__ void pool_div_split_kernel(const float* __restrict__ pool,
                                      const float* __restrict__ cnt,
                                      __bf16* __restrict__ hi,
                                      __bf16* __restrict__ lo)
{
    int i = blockIdx.x * blockDim.x + threadIdx.x;
    if (i < NGRAPH * HID) {
        float v = pool[i] / fmaxf(cnt[i / HID], 1.0f);
        __bf16 h = (__bf16)v;
        hi[i] = h;
        lo[i] = (__bf16)(v - (float)h);
    }
}

// ---------------------------------------------------------------------------
// Host launcher
// ---------------------------------------------------------------------------
extern "C" void kernel_launch(void* const* d_in, const int* in_sizes, int n_in,
                              void* d_out, int out_size, void* d_ws, size_t ws_size,
                              hipStream_t stream)
{
    const float* x     = (const float*)d_in[0];
    const int*   ei    = (const int*)d_in[1];
    const int*   batch = (const int*)d_in[2];
    const float* W1  = (const float*)d_in[3];
    const float* b1  = (const float*)d_in[4];
    const float* W2  = (const float*)d_in[5];
    const float* b2  = (const float*)d_in[6];
    const float* Wf1 = (const float*)d_in[7];
    const float* bf1 = (const float*)d_in[8];
    const float* Wf2 = (const float*)d_in[9];
    const float* bf2 = (const float*)d_in[10];

    const int N = in_sizes[0] / 128;   // 100000
    const int E = in_sizes[1] / 2;     // 1600000
    const int* srci = ei;
    const int* dsti = ei + E;

    // ---- workspace carve-out (256B-aligned slices, ~308 MB total) ----
    char* p = (char*)d_ws;
    auto carve = [&](size_t bytes) -> void* {
        void* r = (void*)p;
        p += (bytes + 255) & ~(size_t)255;
        return r;
    };
    float* H    = (float*)carve((size_t)N * HID * 4);
    float* AGG  = (float*)carve((size_t)N * HID * 4);
    float* DINV = (float*)carve((size_t)N * 4);
    float* POOL = (float*)carve((size_t)NGRAPH * HID * 4);
    float* CNT  = (float*)carve((size_t)NGRAPH * 4);
    float* Z    = (float*)carve((size_t)NGRAPH * HID * 4);
    __bf16* Ahi = (__bf16*)carve((size_t)N * HID * 2);
    __bf16* Alo = (__bf16*)carve((size_t)N * HID * 2);
    __bf16* Bhi = (__bf16*)carve((size_t)HID * HID * 2);
    __bf16* Blo = (__bf16*)carve((size_t)HID * HID * 2);

    // ---- degrees with self-loops -> dinv ----
    init_deg_kernel<<<(N + 255) / 256, 256, 0, stream>>>(DINV, N);
    count_deg_kernel<<<(E + 255) / 256, 256, 0, stream>>>(dsti, DINV, E);
    rsqrt_kernel<<<(N + 255) / 256, 256, 0, stream>>>(DINV, N);

    dim3 gemm_grid((N + TM - 1) / TM, HID / TN);

    // ---- layer 1: H = x @ W1 ----
    split_kernel<<<(N * 128 + 255) / 256, 256, 0, stream>>>(x, Ahi, Alo, N * 128);
    split_t_kernel<<<(128 * HID + 255) / 256, 256, 0, stream>>>(W1, Bhi, Blo, 128, HID);
    wmma_gemm_kernel<<<gemm_grid, 256, 0, stream>>>(Ahi, Alo, Bhi, Blo, nullptr,
                                                    H, N, 128, HID, 0);
    self_init_kernel<<<N, HID, 0, stream>>>(DINV, H, AGG);
    scatter_edges_kernel<<<E, HID, 0, stream>>>(srci, dsti, DINV, H, AGG, E);
    // h1 = relu(AGG + b1), emitted directly as bf16 hi/lo for GEMM2
    bias_relu_split_kernel<<<(N * HID + 255) / 256, 256, 0, stream>>>(
        AGG, b1, Ahi, Alo, N * HID);

    // ---- layer 2: AGG = h1 @ W2 ----
    split_t_kernel<<<(HID * HID + 255) / 256, 256, 0, stream>>>(W2, Bhi, Blo, HID, HID);
    wmma_gemm_kernel<<<gemm_grid, 256, 0, stream>>>(Ahi, Alo, Bhi, Blo, nullptr,
                                                    AGG, N, HID, HID, 0);
    self_init_kernel<<<N, HID, 0, stream>>>(DINV, AGG, H);
    scatter_edges_kernel<<<E, HID, 0, stream>>>(srci, dsti, DINV, AGG, H, E);
    bias_relu_kernel<<<(N * HID + 255) / 256, 256, 0, stream>>>(H, b2, AGG, N * HID);

    // ---- global mean pool ----
    zero_pool_kernel<<<(NGRAPH * HID + 255) / 256, 256, 0, stream>>>(POOL, CNT);
    pool_accum_kernel<<<N, HID, 0, stream>>>(batch, AGG, POOL, CNT);
    pool_div_split_kernel<<<(NGRAPH * HID + 255) / 256, 256, 0, stream>>>(
        POOL, CNT, Ahi, Alo);

    // ---- head: Z = relu(pooled @ Wf1 + bf1) ----
    split_t_kernel<<<(HID * HID + 255) / 256, 256, 0, stream>>>(Wf1, Bhi, Blo, HID, HID);
    dim3 head_grid(1, HID / TN);
    wmma_gemm_kernel<<<head_grid, 256, 0, stream>>>(Ahi, Alo, Bhi, Blo, bf1,
                                                    Z, NGRAPH, HID, HID, 1);

    // ---- out = softplus(Z @ Wf2 + bf2) + 1e-3 ----
    split_kernel<<<(NGRAPH * HID + 255) / 256, 256, 0, stream>>>(Z, Ahi, Alo,
                                                                 NGRAPH * HID);
    split_t_kernel<<<(HID * 10 + 255) / 256, 256, 0, stream>>>(Wf2, Bhi, Blo, HID, 10);
    dim3 out_grid(1, 1);
    wmma_gemm_kernel<<<out_grid, 256, 0, stream>>>(Ahi, Alo, Bhi, Blo, bf2,
                                                   (float*)d_out, NGRAPH, HID, 10, 2);
}
